// AgentAttention1_22900765622467
// MI455X (gfx1250) — matrix operand order repaired and checked by reading, not compiled, not run on hardware
//
#include <hip/hip_runtime.h>
#include <hip/hip_bf16.h>

// ---------------------------------------------------------------------------
// CDNA5 (gfx1250) agent-attention. wave32, WMMA f32_16x16x32_bf16 everywhere.
// Async global->LDS staging (ASYNCcnt) where no dtype conversion is needed.
// ---------------------------------------------------------------------------

typedef __attribute__((ext_vector_type(16))) __bf16 bf16x16;
typedef __attribute__((ext_vector_type(8)))  float  floatx8;

#define M_TOT   100352      // 512 * 196
#define N_TOK   196
#define PAD_N   208         // 13 * 16
#define PAD_NV  224         // 7  * 32
#define CDIM    256
#define HD      32
#define AGN     49
#define PAD_A   64
#define SCALE   0.17677669529663687f   // 1/sqrt(32)

// Async copy of 16 bytes/lane: global -> LDS, tracked by ASYNCcnt.
__device__ __forceinline__ void async_copy_b128(const void* gptr, void* lptr) {
  const unsigned lds = (unsigned)(uintptr_t)lptr;
  asm volatile("global_load_async_to_lds_b128 %0, %1, off"
               :: "v"(lds), "v"(gptr) : "memory");
}
__device__ __forceinline__ void async_wait0() {
  asm volatile("s_wait_asynccnt 0x0" ::: "memory");
}

// ----- fragment loaders (wave32): contiguous k = 16*half + e in both A and B ---------------------

// A 16x32 bf16: lane m = lane&15, half = lane>>4; VGPR r holds K pair
// {2r, 2r+1}, plus 16*half.
__device__ __forceinline__ bf16x16 frag_a(const __bf16* p, int ld, int lane) {
  const int m = lane & 15, half = (lane >> 4) & 1;
  const __bf16* row = p + m * ld + 16 * half;
  bf16x16 a;
#pragma unroll
  for (int r = 0; r < 8; ++r) {
    const int k = 2 * r;
    a[2 * r]     = row[k];
    a[2 * r + 1] = row[k + 1];
  }
  return a;
}

// B 32x16 from a ROW-MAJOR S[n][k] source (B[k][n] = S[n][k]):
// b[e] = S[lane&15][16*half + e]  -> 16 contiguous bf16 per lane.
__device__ __forceinline__ bf16x16 frag_b_rm(const __bf16* p, int ld, int lane) {
  const int nl = lane & 15, half = (lane >> 4) & 1;
  return *(const bf16x16*)(p + nl * ld + 16 * half);
}

// B 32x16 from a COLUMN-LAID S[k][n] source (B[k][n] = S[k][n]):
// b[e] = S[16*half + e][lane&15]  -> strided by ld.
__device__ __forceinline__ bf16x16 frag_b_cm(const __bf16* p, int ld, int lane) {
  const int nl = lane & 15, half = (lane >> 4) & 1;
  const __bf16* src = p + 16 * half * ld + nl;
  bf16x16 b;
#pragma unroll
  for (int e = 0; e < 16; ++e) b[e] = src[e * ld];
  return b;
}

__device__ __forceinline__ floatx8 wmma_bf16(bf16x16 a, bf16x16 b, floatx8 c) {
  return __builtin_amdgcn_wmma_f32_16x16x32_bf16(false, a, false, b,
                                                 (short)0, c, false, false);
}

// ---------------------------------------------------------------------------
// Kernel 1: QKV projection.  C[M=100352, N=768] = bf16(X) @ [Wq;Wkv]^T
// Block 256 thr (8 waves, 4x2), tile 128x64, K-step 32. (fp32->bf16 staging
// must round-trip VGPRs for the conversion, so no async path here.)
// ---------------------------------------------------------------------------
__global__ __launch_bounds__(256) void qkv_gemm_kernel(
    const float* __restrict__ X, const float* __restrict__ Wq,
    const float* __restrict__ Wkv, __bf16* __restrict__ Qw,
    __bf16* __restrict__ Kw, __bf16* __restrict__ Vw) {
  __shared__ __bf16 sA[128 * 32];
  __shared__ __bf16 sB[64 * 32];
  const int tid = threadIdx.x, lane = tid & 31, wv = tid >> 5;
  const int wm = wv >> 1, wn = wv & 1;
  const size_t rowBase = (size_t)blockIdx.x * 128;
  const int nBase = blockIdx.y * 64;

  floatx8 acc[2][2] = {};

  for (int k0 = 0; k0 < CDIM; k0 += 32) {
    // Stage A (128x32 fp32 -> bf16): 1024 float4, 4 per thread.
#pragma unroll
    for (int i = 0; i < 4; ++i) {
      const int q = tid * 4 + i;
      const int r = q >> 3, c4 = q & 7;
      const float4 v = *(const float4*)(X + (rowBase + r) * CDIM + k0 + c4 * 4);
      __bf16* d = sA + r * 32 + c4 * 4;
      d[0] = (__bf16)v.x; d[1] = (__bf16)v.y;
      d[2] = (__bf16)v.z; d[3] = (__bf16)v.w;
    }
    // Stage B rows of [Wq;Wkv] (64x32 fp32 -> bf16): 512 float4, 2 per thread.
#pragma unroll
    for (int i = 0; i < 2; ++i) {
      const int q = tid * 2 + i;
      const int r = q >> 3, c4 = q & 7;
      const int ng = nBase + r;
      const float* wrow = (ng < CDIM) ? (Wq + (size_t)ng * CDIM)
                                      : (Wkv + (size_t)(ng - CDIM) * CDIM);
      const float4 v = *(const float4*)(wrow + k0 + c4 * 4);
      __bf16* d = sB + r * 32 + c4 * 4;
      d[0] = (__bf16)v.x; d[1] = (__bf16)v.y;
      d[2] = (__bf16)v.z; d[3] = (__bf16)v.w;
    }
    // Prefetch next A k-tile (lowers to global_prefetch_b8).
    if (k0 + 32 < CDIM)
      __builtin_prefetch(X + (rowBase + (tid >> 1)) * CDIM + k0 + 32, 0, 1);
    __syncthreads();

    const bf16x16 a0 = frag_a(sA + (wm * 32) * 32, 32, lane);
    const bf16x16 a1 = frag_a(sA + (wm * 32 + 16) * 32, 32, lane);
    const bf16x16 b0 = frag_b_rm(sB + (wn * 32) * 32, 32, lane);
    const bf16x16 b1 = frag_b_rm(sB + (wn * 32 + 16) * 32, 32, lane);
    acc[0][0] = wmma_bf16(a0, b0, acc[0][0]);
    acc[0][1] = wmma_bf16(a0, b1, acc[0][1]);
    acc[1][0] = wmma_bf16(a1, b0, acc[1][0]);
    acc[1][1] = wmma_bf16(a1, b1, acc[1][1]);
    __syncthreads();
  }

  const int nl = lane & 15, half = (lane >> 4) & 1;
#pragma unroll
  for (int i = 0; i < 2; ++i)
#pragma unroll
    for (int j = 0; j < 2; ++j)
#pragma unroll
      for (int e = 0; e < 8; ++e) {
        const size_t row = rowBase + wm * 32 + i * 16 + e + 8 * half;
        const int ng = nBase + wn * 32 + j * 16 + nl;
        const __bf16 v = (__bf16)acc[i][j][e];
        if (ng < 256)      Qw[row * CDIM + ng]       = v;
        else if (ng < 512) Kw[row * CDIM + (ng-256)] = v;
        else               Vw[row * CDIM + (ng-512)] = v;
      }
}

// ---------------------------------------------------------------------------
// Kernel 2: fused per-(batch, head) agent attention, fully LDS-resident.
// Block = 128 threads (4 waves). Grid = 512*8. Q/K/V staged with
// GLOBAL_LOAD_ASYNC_TO_LDS_B128 (ASYNCcnt).
// ---------------------------------------------------------------------------
#define K2_SMEM (( (PAD_N*HD)*2 + (PAD_NV*HD) + (PAD_A*HD)*2 + (PAD_A*PAD_NV) \
                 + (PAD_N*PAD_A) ) * 2 + (PAD_A * PAD_N) * 4)

__global__ __launch_bounds__(128) void attn_kernel(
    const __bf16* __restrict__ Qw, const __bf16* __restrict__ Kw,
    const __bf16* __restrict__ Vw, __bf16* __restrict__ Aw) {
  extern __shared__ char smem[];
  __bf16* sQ  = (__bf16*)smem;            // [208][32]
  __bf16* sK  = sQ  + PAD_N * HD;         // [208][32]
  __bf16* sV  = sK  + PAD_N * HD;         // [224][32]
  __bf16* sAg = sV  + PAD_NV * HD;        // [64][32]
  __bf16* sAV = sAg + PAD_A * HD;         // [64][32]
  __bf16* sP1 = sAV + PAD_A * HD;         // [64][224]
  __bf16* sP2 = sP1 + PAD_A * PAD_NV;     // [208][64]
  float*  sS  = (float*)(sP2 + PAD_N * PAD_A);  // 13312 f32, reused

  const int tid = threadIdx.x, lane = tid & 31, wv = tid >> 5;
  const int nl = lane & 15, half = (lane >> 4) & 1;
  const int bb = blockIdx.x >> 3, h = blockIdx.x & 7;

  const __bf16* Qg = Qw + ((size_t)bb * N_TOK) * CDIM + h * HD;
  const __bf16* Kg = Kw + ((size_t)bb * N_TOK) * CDIM + h * HD;
  const __bf16* Vg = Vw + ((size_t)bb * N_TOK) * CDIM + h * HD;

  // ---- Phase 0: async-stage Q/K/V head slices (16B/lane, 4 lanes/row) ----
  for (int idx = tid; idx < N_TOK * 4; idx += 128) {
    const int row = idx >> 2, c = idx & 3;
    const size_t goff = (size_t)row * CDIM + c * 8;
    const int    loff = row * HD + c * 8;
    async_copy_b128(Qg + goff, sQ + loff);
    async_copy_b128(Kg + goff, sK + loff);
    async_copy_b128(Vg + goff, sV + loff);
  }
  // Zero the WMMA padding rows with ordinary DS stores (covered by barrier).
  for (int i = tid; i < (PAD_N - N_TOK) * HD; i += 128) {
    sQ[N_TOK * HD + i] = (__bf16)0.0f;
    sK[N_TOK * HD + i] = (__bf16)0.0f;
  }
  for (int i = tid; i < (PAD_NV - N_TOK) * HD; i += 128)
    sV[N_TOK * HD + i] = (__bf16)0.0f;
  async_wait0();
  __syncthreads();

  // ---- Phase 1: agent pooling (2x2 mean over 14x14 grid -> 7x7) ----
  for (int idx = tid; idx < PAD_A * HD; idx += 128) {
    const int a = idx >> 5, d = idx & 31;
    float v = 0.f;
    if (a < AGN) {
      const int p1 = a / 7, p2 = a % 7;
      const int n0 = (2 * p1) * 14 + 2 * p2;
      v = 0.25f * ((float)sQ[n0 * HD + d]        + (float)sQ[(n0 + 1) * HD + d] +
                   (float)sQ[(n0 + 14) * HD + d] + (float)sQ[(n0 + 15) * HD + d]);
    }
    sAg[idx] = (__bf16)v;
  }
  __syncthreads();

  // ---- Phase 2: S1 = Agent(64x32) @ K^T -> [64][208] logits ----
  {
    const bf16x16 a = frag_a(sAg + (wv * 16) * HD, HD, lane);
#pragma unroll 1
    for (int tn = 0; tn < 13; ++tn) {
      floatx8 acc = {};
      const bf16x16 b = frag_b_rm(sK + (tn * 16) * HD, HD, lane);
      acc = wmma_bf16(a, b, acc);
#pragma unroll
      for (int e = 0; e < 8; ++e)
        sS[(wv * 16 + e + 8 * half) * PAD_N + tn * 16 + nl] = acc[e];
    }
  }
  __syncthreads();

  // ---- Phase 3: softmax rows of S1 -> P1 bf16 [64][224] (padding = 0) ----
  for (int i = tid; i < PAD_A * PAD_NV; i += 128) {
    const int r = i / PAD_NV, c = i % PAD_NV;
    if (c >= N_TOK || r >= AGN) sP1[i] = (__bf16)0.0f;
  }
  if (tid < AGN) {
    float* row = sS + tid * PAD_N;
    float m = -1e30f;
    for (int n = 0; n < N_TOK; ++n) { const float v = row[n] * SCALE; row[n] = v; m = fmaxf(m, v); }
    float s = 0.f;
    for (int n = 0; n < N_TOK; ++n) { const float e = __expf(row[n] - m); row[n] = e; s += e; }
    const float inv = 1.f / s;
    for (int n = 0; n < N_TOK; ++n) sP1[tid * PAD_NV + n] = (__bf16)(row[n] * inv);
  }
  __syncthreads();

  // ---- Phase 4: agent_v = P1(64x224) @ V(224x32) -> sAV bf16 [64][32] ----
#pragma unroll 1
  for (int tn = 0; tn < 2; ++tn) {
    floatx8 acc = {};
#pragma unroll 1
    for (int kk = 0; kk < 7; ++kk) {
      const bf16x16 a = frag_a(sP1 + (wv * 16) * PAD_NV + kk * 32, PAD_NV, lane);
      const bf16x16 b = frag_b_cm(sV + (kk * 32) * HD + tn * 16, HD, lane);
      acc = wmma_bf16(a, b, acc);
    }
#pragma unroll
    for (int e = 0; e < 8; ++e)
      sAV[(wv * 16 + e + 8 * half) * HD + tn * 16 + nl] = (__bf16)acc[e];
  }
  __syncthreads();

  // ---- Phase 5: S2 = Q(208x32) @ Agent^T -> [208][64] logits ----
#pragma unroll 1
  for (int tm = wv; tm < 13; tm += 4) {
    const bf16x16 a = frag_a(sQ + tm * 16 * HD, HD, lane);
#pragma unroll 1
    for (int tn = 0; tn < 4; ++tn) {
      floatx8 acc = {};
      const bf16x16 b = frag_b_rm(sAg + tn * 16 * HD, HD, lane);
      acc = wmma_bf16(a, b, acc);
#pragma unroll
      for (int e = 0; e < 8; ++e)
        sS[(tm * 16 + e + 8 * half) * PAD_A + tn * 16 + nl] = acc[e];
    }
  }
  __syncthreads();

  // ---- Phase 6: softmax rows of S2 (over 49 agents) -> P2 bf16 [208][64] ----
  for (int i = tid; i < PAD_N * PAD_A; i += 128) {
    const int r = i >> 6, c = i & 63;
    if (c >= AGN || r >= N_TOK) sP2[i] = (__bf16)0.0f;
  }
  for (int n = tid; n < N_TOK; n += 128) {
    float* row = sS + n * PAD_A;
    float m = -1e30f;
    for (int a = 0; a < AGN; ++a) { const float v = row[a] * SCALE; row[a] = v; m = fmaxf(m, v); }
    float s = 0.f;
    for (int a = 0; a < AGN; ++a) { const float e = __expf(row[a] - m); row[a] = e; s += e; }
    const float inv = 1.f / s;
    for (int a = 0; a < AGN; ++a) sP2[n * PAD_A + a] = (__bf16)(row[a] * inv);
  }
  __syncthreads();

  // ---- Phase 7: out = P2(208x64) @ agent_v(64x32) -> Aw bf16 ----
#pragma unroll 1
  for (int tm = wv; tm < 13; tm += 4) {
#pragma unroll 1
    for (int tn = 0; tn < 2; ++tn) {
      floatx8 acc = {};
#pragma unroll
      for (int kk = 0; kk < 2; ++kk) {
        const bf16x16 a = frag_a(sP2 + tm * 16 * PAD_A + kk * 32, PAD_A, lane);
        const bf16x16 b = frag_b_cm(sAV + (kk * 32) * HD + tn * 16, HD, lane);
        acc = wmma_bf16(a, b, acc);
      }
#pragma unroll
      for (int e = 0; e < 8; ++e) {
        const int row = tm * 16 + e + 8 * half;
        if (row < N_TOK)
          Aw[((size_t)bb * N_TOK + row) * CDIM + h * HD + tn * 16 + nl] =
              (__bf16)acc[e];
      }
    }
  }
}

// ---------------------------------------------------------------------------
// Kernel 3: output projection. Out[M,256] = Attn_bf16 @ Wproj^T + bias (fp32)
// A tile is already bf16 -> staged with GLOBAL_LOAD_ASYNC_TO_LDS_B128.
// ---------------------------------------------------------------------------
__global__ __launch_bounds__(256) void proj_gemm_kernel(
    const __bf16* __restrict__ A, const float* __restrict__ Wp,
    const float* __restrict__ bias, float* __restrict__ Out) {
  __shared__ __bf16 sA[128 * 32];
  __shared__ __bf16 sB[64 * 32];
  const int tid = threadIdx.x, lane = tid & 31, wv = tid >> 5;
  const int wm = wv >> 1, wn = wv & 1;
  const size_t rowBase = (size_t)blockIdx.x * 128;
  const int nBase = blockIdx.y * 64;

  floatx8 acc[2][2] = {};

  for (int k0 = 0; k0 < CDIM; k0 += 32) {
    // Stage A (already bf16): async 16B/lane, 512 chunks, 2 per thread.
#pragma unroll
    for (int i = 0; i < 2; ++i) {
      const int q = tid * 2 + i;
      const int r = q >> 2, c = q & 3;
      async_copy_b128(A + (rowBase + r) * CDIM + k0 + c * 8,
                      sA + r * 32 + c * 8);
    }
    // Stage B rows of Wproj (fp32 -> bf16) through VGPRs.
#pragma unroll
    for (int i = 0; i < 2; ++i) {
      const int q = tid * 2 + i;
      const int r = q >> 3, c4 = q & 7;
      const float4 v =
          *(const float4*)(Wp + (size_t)(nBase + r) * CDIM + k0 + c4 * 4);
      __bf16* d = sB + r * 32 + c4 * 4;
      d[0] = (__bf16)v.x; d[1] = (__bf16)v.y;
      d[2] = (__bf16)v.z; d[3] = (__bf16)v.w;
    }
    async_wait0();
    __syncthreads();

    const bf16x16 a0 = frag_a(sA + (wm * 32) * 32, 32, lane);
    const bf16x16 a1 = frag_a(sA + (wm * 32 + 16) * 32, 32, lane);
    const bf16x16 b0 = frag_b_rm(sB + (wn * 32) * 32, 32, lane);
    const bf16x16 b1 = frag_b_rm(sB + (wn * 32 + 16) * 32, 32, lane);
    acc[0][0] = wmma_bf16(a0, b0, acc[0][0]);
    acc[0][1] = wmma_bf16(a0, b1, acc[0][1]);
    acc[1][0] = wmma_bf16(a1, b0, acc[1][0]);
    acc[1][1] = wmma_bf16(a1, b1, acc[1][1]);
    __syncthreads();
  }

  const int nl = lane & 15, half = (lane >> 4) & 1;
#pragma unroll
  for (int i = 0; i < 2; ++i)
#pragma unroll
    for (int j = 0; j < 2; ++j) {
      const int ng = nBase + wn * 32 + j * 16 + nl;
      const float bv = bias[ng];
#pragma unroll
      for (int e = 0; e < 8; ++e) {
        const size_t row = rowBase + wm * 32 + i * 16 + e + 8 * half;
        Out[row * CDIM + ng] = acc[i][j][e] + bv;
      }
    }
}

// ---------------------------------------------------------------------------
extern "C" void kernel_launch(void* const* d_in, const int* in_sizes, int n_in,
                              void* d_out, int out_size, void* d_ws,
                              size_t ws_size, hipStream_t stream) {
  const float* x      = (const float*)d_in[0];
  const float* wq     = (const float*)d_in[1];
  const float* wkv    = (const float*)d_in[2];
  const float* w_proj = (const float*)d_in[3];
  const float* b_proj = (const float*)d_in[4];
  float* out = (float*)d_out;

  const size_t plane = (size_t)M_TOT * CDIM;  // bf16 elements per buffer
  __bf16* Qw = (__bf16*)d_ws;
  __bf16* Kw = Qw + plane;
  __bf16* Vw = Kw + plane;
  __bf16* Aw = Vw + plane;

  (void)hipFuncSetAttribute((const void*)attn_kernel,
                            hipFuncAttributeMaxDynamicSharedMemorySize,
                            K2_SMEM);

  qkv_gemm_kernel<<<dim3(M_TOT / 128, 12), 256, 0, stream>>>(x, wq, wkv, Qw,
                                                             Kw, Vw);
  attn_kernel<<<dim3(512 * 8), 128, K2_SMEM, stream>>>(Qw, Kw, Vw, Aw);
  proj_gemm_kernel<<<dim3(M_TOT / 128, 4), 256, 0, stream>>>(Aw, w_proj,
                                                             b_proj, out);
}
